// FcaBlock_6004364280490
// MI455X (gfx1250) — matrix operand order repaired and checked by reading, not compiled, not run on hardware
//
#include <hip/hip_runtime.h>
#include <cstdint>

typedef _Float16 v16h __attribute__((ext_vector_type(16)));
typedef _Float16 h8   __attribute__((ext_vector_type(8)));
typedef float    v8f  __attribute__((ext_vector_type(8)));

union H16 { v16h v; h8 h[2]; };

__device__ inline v8f wmma_f16(v16h a, v16h b, v8f c) {
    return __builtin_amdgcn_wmma_f32_16x16x32_f16(false, a, false, b, (short)0, c, false, false);
}

__device__ inline float red16_max(float v) {
    v = fmaxf(v, __shfl_xor(v, 1, 32));
    v = fmaxf(v, __shfl_xor(v, 2, 32));
    v = fmaxf(v, __shfl_xor(v, 4, 32));
    v = fmaxf(v, __shfl_xor(v, 8, 32));
    return v;
}
__device__ inline float red16_sum(float v) {
    v += __shfl_xor(v, 1, 32);
    v += __shfl_xor(v, 2, 32);
    v += __shfl_xor(v, 4, 32);
    v += __shfl_xor(v, 8, 32);
    return v;
}

// ---------------------------------------------------------------------------
// Weight convert+transpose: in f32 [R][C] -> out f16 [C][R]  (Bt[k][n]=W[n][k])
// ---------------------------------------------------------------------------
__global__ void convert_transpose_kernel(const float* __restrict__ in,
                                         _Float16* __restrict__ out, int R, int C) {
    int i = blockIdx.x * 256 + threadIdx.x;
    if (i >= R * C) return;
    int r = i / C, c = i - r * C;
    out[(size_t)c * R + r] = (_Float16)in[(size_t)r * C + c];
}

// ---------------------------------------------------------------------------
// LayerNorm: 1 block / row (C=384), optional per-row scale, f16 output
// ---------------------------------------------------------------------------
__global__ __launch_bounds__(128) void ln_kernel(
    const float* __restrict__ in, const float* __restrict__ rowscale, int rs_mod,
    const float* __restrict__ g, const float* __restrict__ bta,
    _Float16* __restrict__ out, int out_row_off) {
    __shared__ float rs_[128], rq_[128];
    int row = blockIdx.x, t = threadIdx.x;
    const float* src = in + (size_t)row * 384;
    float sc = rowscale ? rowscale[row % rs_mod] : 1.0f;
    float v0 = src[t] * sc, v1 = src[t + 128] * sc, v2 = src[t + 256] * sc;
    rs_[t] = v0 + v1 + v2;
    rq_[t] = v0 * v0 + v1 * v1 + v2 * v2;
    __syncthreads();
    for (int o = 64; o > 0; o >>= 1) {
        if (t < o) { rs_[t] += rs_[t + o]; rq_[t] += rq_[t + o]; }
        __syncthreads();
    }
    float mean = rs_[0] * (1.0f / 384.0f);
    float var  = rq_[0] * (1.0f / 384.0f) - mean * mean;
    float r = rsqrtf(var + 1e-5f);
    _Float16* dst = out + (size_t)(row + out_row_off) * 384;
    dst[t]       = (_Float16)((v0 - mean) * r * g[t]       + bta[t]);
    dst[t + 128] = (_Float16)((v1 - mean) * r * g[t + 128] + bta[t + 128]);
    dst[t + 256] = (_Float16)((v2 - mean) * r * g[t + 256] + bta[t + 256]);
}

// LN(LN(x)) fused: norm2 then mlp_norm, f16 output
__global__ __launch_bounds__(128) void dual_ln_kernel(
    const float* __restrict__ in,
    const float* __restrict__ g1, const float* __restrict__ b1,
    const float* __restrict__ g2, const float* __restrict__ b2,
    _Float16* __restrict__ out) {
    __shared__ float rs_[128], rq_[128];
    int row = blockIdx.x, t = threadIdx.x;
    const float* src = in + (size_t)row * 384;
    float v[3] = {src[t], src[t + 128], src[t + 256]};
    rs_[t] = v[0] + v[1] + v[2];
    rq_[t] = v[0] * v[0] + v[1] * v[1] + v[2] * v[2];
    __syncthreads();
    for (int o = 64; o > 0; o >>= 1) {
        if (t < o) { rs_[t] += rs_[t + o]; rq_[t] += rq_[t + o]; }
        __syncthreads();
    }
    float mean = rs_[0] * (1.0f / 384.0f);
    float var  = rq_[0] * (1.0f / 384.0f) - mean * mean;
    float r = rsqrtf(var + 1e-5f);
    float y[3];
    for (int i = 0; i < 3; ++i) {
        int c = t + i * 128;
        y[i] = (v[i] - mean) * r * g1[c] + b1[c];
    }
    __syncthreads();
    rs_[t] = y[0] + y[1] + y[2];
    rq_[t] = y[0] * y[0] + y[1] * y[1] + y[2] * y[2];
    __syncthreads();
    for (int o = 64; o > 0; o >>= 1) {
        if (t < o) { rs_[t] += rs_[t + o]; rq_[t] += rq_[t + o]; }
        __syncthreads();
    }
    float mean2 = rs_[0] * (1.0f / 384.0f);
    float var2  = rq_[0] * (1.0f / 384.0f) - mean2 * mean2;
    float r2 = rsqrtf(var2 + 1e-5f);
    _Float16* dst = out + (size_t)row * 384;
    for (int i = 0; i < 3; ++i) {
        int c = t + i * 128;
        dst[c] = (_Float16)((y[i] - mean2) * r2 * g2[c] + b2[c]);
    }
}

// ---------------------------------------------------------------------------
// Generic WMMA GEMM: C[M,N] = A[M,K] @ Bt[K,N]  (Bt = W^T pre-transposed f16)
// 256 threads / 8 waves; block tile 128x64; wave tile 32x32 (4 wmma / K-step)
// Fused epilogues: 0=QKV scatter(+SCALE on q), 1=residual(out=res+gamma*v), 2=GELU f16
// ---------------------------------------------------------------------------
struct GemmParams {
    const _Float16* A; const _Float16* Bt; const float* bias;
    int M, N, K, mode;
    _Float16 *qp, *kp, *vp;               // mode 0
    float* out_f32; const float* resid; const float* gamma;  // mode 1
    _Float16* out_f16;                    // mode 2
};

__global__ __launch_bounds__(256) void gemm_kernel(GemmParams p) {
    __shared__ __align__(16) _Float16 As[128 * 40];  // [m][k], padded stride 40
    __shared__ __align__(16) _Float16 Bs[32 * 72];   // [k][n], padded stride 72
    int tiles_n = p.N >> 6;
    int tm = blockIdx.x / tiles_n, tn = blockIdx.x - tm * tiles_n;
    int tid = threadIdx.x, lane = tid & 31, wid = tid >> 5;
    int lhalf = lane >> 4, l15 = lane & 15;
    int wm = wid >> 1, wn = wid & 1;
    int m0 = tm * 128, n0 = tn * 64;
    v8f acc[2][2] = {};
    int ksteps = p.K >> 5;
    for (int ks = 0; ks < ksteps; ++ks) {
        int k0 = ks << 5;
        __syncthreads();
        {   // A tile: 128x32 halves, 32 halves per thread-pair row
            int r = tid >> 1, part = tid & 1;
            const uint4* src = (const uint4*)(p.A + (size_t)(m0 + r) * p.K + k0 + part * 16);
            uint4 d0 = src[0], d1 = src[1];
            uint4* dst = (uint4*)(As + r * 40 + part * 16);
            dst[0] = d0; dst[1] = d1;
        }
        {   // B tile: 32x64 halves
            int kk = tid >> 3, part = tid & 7;
            const uint4* src = (const uint4*)(p.Bt + (size_t)(k0 + kk) * p.N + n0 + part * 8);
            uint4* dst = (uint4*)(Bs + kk * 72 + part * 8);
            dst[0] = src[0];
        }
        __syncthreads();
        H16 a[2], b[2];
        for (int i = 0; i < 2; ++i) {   // A frag: lane=M, slots: K=(s&7)+8*lhalf+16*(s>>3)
            const _Float16* ap = As + (wm * 32 + i * 16 + l15) * 40 + 8 * lhalf;
            a[i].h[0] = *(const h8*)ap;
            a[i].h[1] = *(const h8*)(ap + 16);
        }
        for (int j = 0; j < 2; ++j) {   // B frag: lane=N, slots: K=(s&7)+8*lhalf+16*(s>>3)
            const _Float16* bp = Bs + wn * 32 + j * 16 + l15;
            for (int s = 0; s < 16; ++s)
                b[j].v[s] = bp[((s & 7) + 8 * lhalf + 16 * (s >> 3)) * 72];
        }
        for (int i = 0; i < 2; ++i)
            for (int j = 0; j < 2; ++j)
                acc[i][j] = wmma_f16(a[i].v, b[j].v, acc[i][j]);
    }
    // Epilogue. D layout: lane=N (l15), vgpr v -> M = v + 8*lhalf
    for (int i = 0; i < 2; ++i)
        for (int j = 0; j < 2; ++j)
            for (int v = 0; v < 8; ++v) {
                int mg = m0 + wm * 32 + i * 16 + v + 8 * lhalf;
                int ng = n0 + wn * 32 + j * 16 + l15;
                float val = acc[i][j][v] + p.bias[ng];
                if (p.mode == 0) {
                    int sec = ng / 384, rem = ng - sec * 384;
                    int hh = rem >> 5, hd = rem & 31;
                    if (sec == 0) {
                        if (mg < 8192) {
                            int bb = mg >> 10, n = mg & 1023;
                            p.qp[(((size_t)bb * 12 + hh) * 1024 + n) * 32 + hd] =
                                (_Float16)(val * 0.17677669529663687f);
                        }
                    } else {
                        int bb, slot;
                        if (mg < 8192) { bb = mg >> 10; slot = mg & 1023; }
                        else { bb = (mg - 8192) >> 7; slot = 1024 + ((mg - 8192) & 127); }
                        _Float16* dst = (sec == 1) ? p.kp : p.vp;
                        dst[(((size_t)bb * 12 + hh) * 1152 + slot) * 32 + hd] = (_Float16)val;
                    }
                } else if (p.mode == 1) {
                    size_t o = (size_t)mg * p.N + ng;
                    p.out_f32[o] = p.resid[o] + p.gamma[ng] * val;
                } else {  // GELU (exact, erf)
                    float gv = 0.5f * val * (1.0f + erff(val * 0.70710678118654752f));
                    p.out_f16[(size_t)mg * p.N + ng] = (_Float16)gv;
                }
            }
}

// ---------------------------------------------------------------------------
// Flash-style attention. Block = 4 waves = (b,h,64 q rows); wave = 16 rows.
// 36 iters x 32 keys: 2 score WMMAs + online softmax + 2 P@V WMMAs.
// ---------------------------------------------------------------------------
__global__ __launch_bounds__(128) void attn_kernel(
    const _Float16* __restrict__ qh, const _Float16* __restrict__ kh,
    const _Float16* __restrict__ vh,
    const float* __restrict__ abias,   // [12][1024]
    const int*   __restrict__ bidx,    // [1024][1024]
    const float* __restrict__ sbias,   // [12][128]
    _Float16* __restrict__ aout) {     // [8][1024][384], c = h*32+hd
    __shared__ __align__(16) _Float16 KT[32][40];      // [hd][key]  (transposed)
    __shared__ __align__(16) _Float16 VT[32][40];      // [key][hd]
    __shared__ __align__(16) _Float16 PS[4][16][40];   // per-wave P: [m][key 0..31]
    int blk = blockIdx.x;
    int rt = blk & 15, h = (blk >> 4) % 12, b = blk / 192;
    int tid = threadIdx.x, lane = tid & 31, wid = tid >> 5;
    int lhalf = lane >> 4, l15 = lane & 15;
    int row0 = rt * 64 + wid * 16;
    const _Float16* qb = qh + (size_t)(b * 12 + h) * 1024 * 32;
    const _Float16* kb = kh + (size_t)(b * 12 + h) * 1152 * 32;
    const _Float16* vb = vh + (size_t)(b * 12 + h) * 1152 * 32;

    H16 aq;  // q A-frag, loaded once (K = HD = 32)
    {
        const _Float16* qr = qb + (size_t)(row0 + l15) * 32 + 8 * lhalf;
        aq.h[0] = *(const h8*)qr;
        aq.h[1] = *(const h8*)(qr + 16);
    }
    v8f o0 = {}, o1 = {};
    float mi[8], li[8];
    for (int v = 0; v < 8; ++v) { mi[v] = -1e30f; li[v] = 0.0f; }

    for (int it = 0; it < 36; ++it) {
        int kb0 = it * 32;
        __syncthreads();
        {   // cooperative stage of K (transposed) and V tiles (32 keys x 32 hd)
            int key = tid >> 2, part = tid & 3;
            h8 kd = *(const h8*)(kb + (size_t)(kb0 + key) * 32 + part * 8);
            h8 vd = *(const h8*)(vb + (size_t)(kb0 + key) * 32 + part * 8);
            for (int e = 0; e < 8; ++e) KT[part * 8 + e][key] = kd[e];
            *(h8*)&VT[key][part * 8] = vd;
        }
        __syncthreads();
        H16 bk0, bk1;                     // B frags for S = q @ k^T (lane = key)
        for (int s = 0; s < 16; ++s) {
            int kk = (s & 7) + 8 * lhalf + 16 * (s >> 3);
            bk0.v[s] = KT[kk][l15]; bk1.v[s] = KT[kk][16 + l15];
        }
        v8f z = {};
        v8f s0 = wmma_f16(aq.v, bk0.v, z);
        v8f s1 = wmma_f16(aq.v, bk1.v, z);
        for (int v = 0; v < 8; ++v) {
            int m = row0 + v + 8 * lhalf;
            int key0 = kb0 + l15, key1 = key0 + 16;
            float bi0 = (key0 < 1024) ? abias[h * 1024 + bidx[(size_t)m * 1024 + key0]]
                                      : sbias[h * 128 + (key0 - 1024)];
            float bi1 = (key1 < 1024) ? abias[h * 1024 + bidx[(size_t)m * 1024 + key1]]
                                      : sbias[h * 128 + (key1 - 1024)];
            float x0 = s0[v] + bi0, x1v = s1[v] + bi1;
            float tmax = red16_max(fmaxf(x0, x1v));
            float nm = fmaxf(mi[v], tmax);
            float sc = __expf(mi[v] - nm);
            float p0 = __expf(x0 - nm), p1 = __expf(x1v - nm);
            float psum = red16_sum(p0 + p1);
            li[v] = li[v] * sc + psum;
            mi[v] = nm;
            o0[v] *= sc; o1[v] *= sc;
            PS[wid][v + 8 * lhalf][l15]      = (_Float16)p0;   // D->LDS
            PS[wid][v + 8 * lhalf][16 + l15] = (_Float16)p1;
        }
        H16 pa;  // reload P as A-frag (wave-local LDS, in-order DS ops)
        {
            const _Float16* pp = &PS[wid][l15][8 * lhalf];
            pa.h[0] = *(const h8*)pp;
            pa.h[1] = *(const h8*)(pp + 16);
        }
        H16 bv0, bv1;                     // B frags for P @ V (lane = hd)
        for (int s = 0; s < 16; ++s) {
            int kk = (s & 7) + 8 * lhalf + 16 * (s >> 3);
            bv0.v[s] = VT[kk][l15]; bv1.v[s] = VT[kk][16 + l15];
        }
        o0 = wmma_f16(pa.v, bv0.v, o0);
        o1 = wmma_f16(pa.v, bv1.v, o1);
    }
    for (int v = 0; v < 8; ++v) {
        float inv = 1.0f / li[v];
        int n = row0 + v + 8 * lhalf;
        size_t base = ((size_t)b * 1024 + n) * 384 + h * 32;
        aout[base + l15]      = (_Float16)(o0[v] * inv);
        aout[base + 16 + l15] = (_Float16)(o1[v] * inv);
    }
}

// ---------------------------------------------------------------------------
// depthwise 7x7 stride-4 pad-3 on x1 [B,N,C] -> s_up [B,64,C]
// ---------------------------------------------------------------------------
__global__ void dwconv7_kernel(const float* __restrict__ x1, const float* __restrict__ w,
                               float* __restrict__ out) {
    int i = blockIdx.x * 256 + threadIdx.x;  // 8*64*384
    int c = i % 384, rest = i / 384;
    int p = rest & 63, b = rest >> 6;
    int oy = p >> 3, ox = p & 7;
    const float* wc = w + c * 49;
    const float* xb = x1 + (size_t)b * 1024 * 384;
    float acc = 0.0f;
    for (int ky = 0; ky < 7; ++ky) {
        int y = oy * 4 - 3 + ky;
        if ((unsigned)y >= 32u) continue;
        for (int kx = 0; kx < 7; ++kx) {
            int xq = ox * 4 - 3 + kx;
            if ((unsigned)xq >= 32u) continue;
            acc += xb[(size_t)(y * 32 + xq) * 384 + c] * wc[ky * 7 + kx];
        }
    }
    out[i] = acc;
}

// depthwise 3x3 pad-1 + fused BatchNorm (eval) -> x2 [B,N,C]
__global__ void lc3bn_kernel(const float* __restrict__ x1, const float* __restrict__ w,
                             const float* __restrict__ bnw, const float* __restrict__ bnb,
                             const float* __restrict__ bnm, const float* __restrict__ bnv,
                             float* __restrict__ x2) {
    int i = blockIdx.x * 256 + threadIdx.x;  // 8*1024*384
    int c = i % 384, rest = i / 384;
    int n = rest & 1023, b = rest >> 10;
    int y = n >> 5, xx = n & 31;
    const float* wc = w + c * 9;
    const float* xb = x1 + (size_t)b * 1024 * 384;
    float acc = 0.0f;
    for (int ky = 0; ky < 3; ++ky) {
        int yy = y - 1 + ky;
        if ((unsigned)yy >= 32u) continue;
        for (int kx = 0; kx < 3; ++kx) {
            int xq = xx - 1 + kx;
            if ((unsigned)xq >= 32u) continue;
            acc += xb[(size_t)(yy * 32 + xq) * 384 + c] * wc[ky * 3 + kx];
        }
    }
    float scl = bnw[c] * rsqrtf(bnv[c] + 1e-5f);
    x2[i] = (acc - bnm[c]) * scl + bnb[c];
}

// ---------------------------------------------------------------------------
extern "C" void kernel_launch(void* const* d_in, const int* in_sizes, int n_in,
                              void* d_out, int out_size, void* d_ws, size_t ws_size,
                              hipStream_t stream) {
    const float* x        = (const float*)d_in[0];
    const float* s        = (const float*)d_in[1];
    const float* s_scale  = (const float*)d_in[2];
    const float* norm1_g  = (const float*)d_in[3];
    const float* norm1_b  = (const float*)d_in[4];
    const float* s_norm_g = (const float*)d_in[5];
    const float* s_norm_b = (const float*)d_in[6];
    const float* qkv_w    = (const float*)d_in[7];
    const float* qkv_b    = (const float*)d_in[8];
    const float* attn_bi  = (const float*)d_in[9];
    const float* s_tok_b  = (const float*)d_in[10];
    const float* proj_w   = (const float*)d_in[11];
    const float* proj_b   = (const float*)d_in[12];
    const float* gamma1   = (const float*)d_in[13];
    const float* dw_w     = (const float*)d_in[14];
    const float* lc_w     = (const float*)d_in[15];
    const float* bn_w     = (const float*)d_in[16];
    const float* bn_b     = (const float*)d_in[17];
    const float* bn_mean  = (const float*)d_in[18];
    const float* bn_var   = (const float*)d_in[19];
    const float* norm2_g  = (const float*)d_in[20];
    const float* norm2_b  = (const float*)d_in[21];
    const float* mlpn_g   = (const float*)d_in[22];
    const float* mlpn_b   = (const float*)d_in[23];
    const float* fc1_w    = (const float*)d_in[24];
    const float* fc1_b    = (const float*)d_in[25];
    const float* fc2_w    = (const float*)d_in[26];
    const float* fc2_b    = (const float*)d_in[27];
    const float* gamma2   = (const float*)d_in[28];
    const int*   bias_idx = (const int*)d_in[29];

    char* ws = (char*)d_ws;
    size_t off = 0;
    auto alloc = [&](size_t bytes) {
        size_t o = off; off = (off + bytes + 255) & ~(size_t)255; return o;
    };
    _Float16* wqkvT  = (_Float16*)(ws + alloc((size_t)384 * 1152 * 2));
    _Float16* wprojT = (_Float16*)(ws + alloc((size_t)384 * 384 * 2));
    _Float16* wfc1T  = (_Float16*)(ws + alloc((size_t)384 * 1536 * 2));
    _Float16* wfc2T  = (_Float16*)(ws + alloc((size_t)1536 * 384 * 2));
    _Float16* xn_h   = (_Float16*)(ws + alloc((size_t)9216 * 384 * 2));
    _Float16* q_h    = (_Float16*)(ws + alloc((size_t)8 * 12 * 1024 * 32 * 2));
    _Float16* k_h    = (_Float16*)(ws + alloc((size_t)8 * 12 * 1152 * 32 * 2));
    _Float16* v_h    = (_Float16*)(ws + alloc((size_t)8 * 12 * 1152 * 32 * 2));
    _Float16* attn_o = (_Float16*)(ws + alloc((size_t)8192 * 384 * 2));
    float*    x1     = (float*)   (ws + alloc((size_t)8192 * 384 * 4));
    float*    x2     = (float*)   (ws + alloc((size_t)8192 * 384 * 4));
    // aliased reuse (earlier buffers dead by the time these are live):
    _Float16* h0   = xn_h;                 // 6.3 MB into 7.1 MB span
    _Float16* hmid = q_h;                  // 25.2 MB into q+k+v+attn_o span (26.7 MB)

    // 1) weight transposes to f16
    convert_transpose_kernel<<<(1152 * 384 + 255) / 256, 256, 0, stream>>>(qkv_w, wqkvT, 1152, 384);
    convert_transpose_kernel<<<(384 * 384 + 255) / 256, 256, 0, stream>>>(proj_w, wprojT, 384, 384);
    convert_transpose_kernel<<<(1536 * 384 + 255) / 256, 256, 0, stream>>>(fc1_w, wfc1T, 1536, 384);
    convert_transpose_kernel<<<(384 * 1536 + 255) / 256, 256, 0, stream>>>(fc2_w, wfc2T, 384, 1536);

    // 2) LayerNorms -> xn_h rows [0,8192) x-tokens, [8192,9216) s-tokens (s*s_scale)
    ln_kernel<<<8192, 128, 0, stream>>>(x, nullptr, 1, norm1_g, norm1_b, xn_h, 0);
    ln_kernel<<<1024, 128, 0, stream>>>(s, s_scale, 128, s_norm_g, s_norm_b, xn_h, 8192);

    // 3) fused QKV GEMM 9216x1152x384, scatter into q/k/v (q pre-scaled)
    {
        GemmParams p = {};
        p.A = xn_h; p.Bt = wqkvT; p.bias = qkv_b;
        p.M = 9216; p.N = 1152; p.K = 384; p.mode = 0;
        p.qp = q_h; p.kp = k_h; p.vp = v_h;
        gemm_kernel<<<(9216 / 128) * (1152 / 64), 256, 0, stream>>>(p);
    }

    // 4) flash attention with relative-position + s-token bias
    attn_kernel<<<8 * 12 * 16, 128, 0, stream>>>(q_h, k_h, v_h, attn_bi, bias_idx, s_tok_b, attn_o);

    // 5) proj GEMM + residual: x1 = x + gamma1 * (attn_o @ Wp^T + b)
    {
        GemmParams p = {};
        p.A = attn_o; p.Bt = wprojT; p.bias = proj_b;
        p.M = 8192; p.N = 384; p.K = 384; p.mode = 1;
        p.out_f32 = x1; p.resid = x; p.gamma = gamma1;
        gemm_kernel<<<(8192 / 128) * (384 / 64), 256, 0, stream>>>(p);
    }

    // 6) s_up branch -> d_out[3145728 ..)
    float* out_x  = (float*)d_out;
    float* out_su = out_x + (size_t)8 * 1024 * 384;
    dwconv7_kernel<<<(8 * 64 * 384) / 256, 256, 0, stream>>>(x1, dw_w, out_su);

    // 7) local 3x3 conv + BN -> x2
    lc3bn_kernel<<<(8 * 1024 * 384) / 256, 256, 0, stream>>>(x1, lc_w, bn_w, bn_b, bn_mean, bn_var, x2);

    // 8) LN(LN(x2)) -> h0 f16
    dual_ln_kernel<<<8192, 128, 0, stream>>>(x2, norm2_g, norm2_b, mlpn_g, mlpn_b, h0);

    // 9) fc1 + exact GELU -> hmid f16
    {
        GemmParams p = {};
        p.A = h0; p.Bt = wfc1T; p.bias = fc1_b;
        p.M = 8192; p.N = 1536; p.K = 384; p.mode = 2;
        p.out_f16 = hmid;
        gemm_kernel<<<(8192 / 128) * (1536 / 64), 256, 0, stream>>>(p);
    }

    // 10) fc2 + residual: out_x = x2 + gamma2 * (hmid @ Wfc2^T + b)
    {
        GemmParams p = {};
        p.A = hmid; p.Bt = wfc2T; p.bias = fc2_b;
        p.M = 8192; p.N = 384; p.K = 1536; p.mode = 1;
        p.out_f32 = out_x; p.resid = x2; p.gamma = gamma2;
        gemm_kernel<<<(8192 / 128) * (384 / 64), 256, 0, stream>>>(p);
    }
    (void)in_sizes; (void)n_in; (void)out_size; (void)ws_size;
}
